// MoeAttentionRelativeTransformerEncoderLayer_62869731279206
// MI455X (gfx1250) — matrix operand order repaired and checked by reading, not compiled, not run on hardware
//
#include <hip/hip_runtime.h>

// ---------------------------------------------------------------------------
// Types & helpers for CDNA5 (gfx1250) bf16 WMMA, wave32.
// ---------------------------------------------------------------------------
typedef __attribute__((ext_vector_type(16))) __bf16 bf16x16;
typedef __attribute__((ext_vector_type(8)))  float  f32x8;

__device__ inline f32x8 zero8() {
  f32x8 z;
#pragma unroll
  for (int i = 0; i < 8; ++i) z[i] = 0.0f;
  return z;
}

// native bf16 convert (v_cvt, RNE)
__device__ inline unsigned short f2bf(float f) {
  return __builtin_bit_cast(unsigned short, (__bf16)f);
}
__device__ inline float bf2f(unsigned short s) {
  unsigned int u = ((unsigned int)s) << 16;
  return __builtin_bit_cast(float, u);
}

union US4 { unsigned short s[4]; uint2 u; };

union FragU { bf16x16 v; uint4 q[2]; };

// A-frag 16x32 bf16 from LDS (row-major, rowStride ushorts, 16B aligned rows).
__device__ inline bf16x16 load_a_frag(const unsigned short* lds, int rowStride) {
  int lane = threadIdx.x & 31;
  int m = lane & 15;
  int hi = lane >> 4;
  const unsigned short* p = lds + m * rowStride;
  FragU u;
  u.q[0] = *(const uint4*)(p + (hi ? 16 : 0));
  u.q[1] = *(const uint4*)(p + (hi ? 24 : 8));
  return u.v;
}

// B-frag 32x16 bf16 from transposed LDS tile ldsBT[n*rowStride + k].
__device__ inline bf16x16 load_bt_frag(const unsigned short* lds, int rowStride, int k0) {
  int lane = threadIdx.x & 31;
  int n = lane & 15;
  int hi = lane >> 4;
  const unsigned short* p = lds + n * rowStride + k0 + (hi ? 16 : 0);
  FragU u;
  u.q[0] = *(const uint4*)(p);
  u.q[1] = *(const uint4*)(p + 8);
  return u.v;
}

__device__ inline f32x8 wmma_bf16(bf16x16 a, bf16x16 b, f32x8 c) {
  return __builtin_amdgcn_wmma_f32_16x16x32_bf16(false, a, false, b, (short)0, c,
                                                 false, false);
}

// Problem constants
#define Dm   1024
#define Hh   16
#define Pp   64
#define Ee   8
#define Ff   4096
#define Bb   2
#define Ss   2048
#define Mtok 4096            // B*S

// ---------------------------------------------------------------------------
// 0) f32 -> bf16 bulk convert (x pre-conversion)
// ---------------------------------------------------------------------------
__global__ void tobf_kernel(const float* __restrict__ in,
                            unsigned short* __restrict__ out) {
  int idx = blockIdx.x * 256 + threadIdx.x;          // chunk of 4
  float4 v = *(const float4*)&in[idx * 4];
  US4 r;
  r.s[0] = f2bf(v.x); r.s[1] = f2bf(v.y); r.s[2] = f2bf(v.z); r.s[3] = f2bf(v.w);
  *(uint2*)&out[idx * 4] = r.u;
}

// ---------------------------------------------------------------------------
// 1) Gate logits: logits[which][b][h][s][e] = x[b,s,:] . Wsel[h,e,:]
// ---------------------------------------------------------------------------
__global__ void gate_logits_kernel(const float* __restrict__ x,
                                   const float* __restrict__ wselv,
                                   const float* __restrict__ wselo,
                                   float* __restrict__ logits) {
  __shared__ __align__(16) float xr[Dm];
  int t = blockIdx.x;                       // token 0..4095
  ((float4*)xr)[threadIdx.x] = ((const float4*)x)[t * (Dm / 4) + threadIdx.x];
  __syncthreads();
  int tid = threadIdx.x;
  int which = tid >> 7;                     // 0: v-gate, 1: o-gate
  int he = tid & 127;                       // h*8+e
  const float* wp = (which ? wselo : wselv) + he * Dm;
  float acc = 0.0f;
  for (int k = 0; k < Dm; k += 4) {
    float4 a = *(const float4*)&xr[k];
    float4 b = *(const float4*)&wp[k];
    acc += a.x * b.x + a.y * b.y + a.z * b.z + a.w * b.w;
  }
  int b = t >> 11, s = t & (Ss - 1);
  int h = he >> 3, e = he & 7;
  logits[((((which * Bb) + b) * Hh + h) * Ss + s) * Ee + e] = acc;
}

// ---------------------------------------------------------------------------
// 2) sigmoid + top-2-of-8 mask -> gv / go
// ---------------------------------------------------------------------------
__global__ void gate_topk_kernel(const float* __restrict__ logits,
                                 float* __restrict__ gv, float* __restrict__ go) {
  int idx = blockIdx.x * blockDim.x + threadIdx.x;   // 2*B*H*S
  if (idx >= 2 * Bb * Hh * Ss) return;
  int which = idx / (Bb * Hh * Ss);
  int rem = idx % (Bb * Hh * Ss);
  const float* lp = logits + (which * (Bb * Hh * Ss) + rem) * Ee;
  float sv[8];
#pragma unroll
  for (int e = 0; e < 8; ++e) sv[e] = 1.0f / (1.0f + __expf(-lp[e]));
  int i1 = 0;
#pragma unroll
  for (int e = 1; e < 8; ++e) if (sv[e] > sv[i1]) i1 = e;
  int i2 = (i1 == 0) ? 1 : 0;
#pragma unroll
  for (int e = 0; e < 8; ++e) if (e != i1 && sv[e] > sv[i2]) i2 = e;
  float* out = (which ? go : gv) + rem * Ee;
#pragma unroll
  for (int e = 0; e < 8; ++e) out[e] = (e == i1 || e == i2) ? sv[e] : 0.0f;
}

// ---------------------------------------------------------------------------
// 3) Q/K projection + RoPE.  grid (32 mblocks, 16 h, 2 q/k), 256 thr.
//    Tile 128Mx64N, K=1024, double-buffered software pipeline.
// ---------------------------------------------------------------------------
__global__ void qk_rope_kernel(const unsigned short* __restrict__ xbf,
                               const float* __restrict__ Wq,
                               const float* __restrict__ Wk,
                               unsigned short* __restrict__ qbf,
                               unsigned short* __restrict__ kbf) {
  const int mb = blockIdx.x, h = blockIdx.y, isK = blockIdx.z;
  const float* W = (isK ? Wk : Wq) + h * (Dm * Pp);
  unsigned short* out = isK ? kbf : qbf;
  __shared__ __align__(16) unsigned short ldsA[2][128 * 40];
  __shared__ __align__(16) unsigned short ldsB[2][64 * 40];
  const int tid = threadIdx.x, w = tid >> 5, lane = tid & 31;
  const int ln16 = lane & 15, hi = lane >> 4;
  const int m0 = mb * 128;
  f32x8 acc[4];
#pragma unroll
  for (int i = 0; i < 4; ++i) acc[i] = zero8();

  uint2 ar[4];
  float4 br[2];
  auto gload = [&](int ks) {
#pragma unroll
    for (int it = 0; it < 4; ++it) {                  // A 128x32 (bf16 copy)
      int idx = tid + it * 256;
      int r = idx >> 3, c4 = (idx & 7) * 4;
      ar[it] = *(const uint2*)&xbf[(m0 + r) * Dm + ks + c4];
    }
#pragma unroll
    for (int it = 0; it < 2; ++it) {                  // B 32x64
      int idx = tid + it * 256;
      int k = idx >> 4, n4 = (idx & 15) * 4;
      br[it] = *(const float4*)&W[(ks + k) * Pp + n4];
    }
  };
  auto sstore = [&](int buf) {
#pragma unroll
    for (int it = 0; it < 4; ++it) {
      int idx = tid + it * 256;
      int r = idx >> 3, c4 = (idx & 7) * 4;
      *(uint2*)&ldsA[buf][r * 40 + c4] = ar[it];
    }
#pragma unroll
    for (int it = 0; it < 2; ++it) {
      int idx = tid + it * 256;
      int k = idx >> 4, n4 = (idx & 15) * 4;
      float4 v = br[it];
      ldsB[buf][(n4 + 0) * 40 + k] = f2bf(v.x);
      ldsB[buf][(n4 + 1) * 40 + k] = f2bf(v.y);
      ldsB[buf][(n4 + 2) * 40 + k] = f2bf(v.z);
      ldsB[buf][(n4 + 3) * 40 + k] = f2bf(v.w);
    }
  };

  gload(0);
  int cur = 0;
  for (int ks = 0; ks < Dm; ks += 32) {
    sstore(cur);
    __syncthreads();
    if (ks + 32 < Dm) gload(ks + 32);                 // overlap with compute
    bf16x16 a = load_a_frag(&ldsA[cur][(w * 16) * 40], 40);
#pragma unroll
    for (int i = 0; i < 4; ++i) {
      bf16x16 b = load_bt_frag(&ldsB[cur][(i * 16) * 40], 40, 0);
      acc[i] = wmma_bf16(a, b, acc[i]);
    }
    cur ^= 1;
  }
  // RoPE: 10000^(-i/16) = exp2(-i*log2(10000)/16)
  float inv = exp2f((float)ln16 * -0.83048202372184058696f);
#pragma unroll
  for (int j = 0; j < 8; ++j) {
    int m = m0 + w * 16 + j + hi * 8;
    int s = m & (Ss - 1);
    float ang = (float)s * inv;
    float c = __cosf(ang), sn = __sinf(ang);
    float x1 = acc[0][j], x2 = acc[1][j];
    acc[0][j] = x1 * c - x2 * sn;
    acc[1][j] = x1 * sn + x2 * c;
  }
#pragma unroll
  for (int i = 0; i < 4; ++i)
#pragma unroll
    for (int j = 0; j < 8; ++j) {
      int m = m0 + w * 16 + j + hi * 8;
      int b = m >> 11, s = m & (Ss - 1);
      int p = i * 16 + ln16;
      out[((b * Hh + h) * Ss + s) * Pp + p] = f2bf(acc[i][j]);
    }
}

// ---------------------------------------------------------------------------
// 4) V experts: v[b,h,s,p] = sum_e gv[b,h,s,e] * (x . Wv[h,e])
//    grid (32 tblocks, 16 h).  Tile 128Mx64N, pipelined over flattened (e,ks).
// ---------------------------------------------------------------------------
__global__ void vexpert_kernel(const unsigned short* __restrict__ xbf,
                               const float* __restrict__ Wv,
                               const float* __restrict__ gv,
                               unsigned short* __restrict__ vbf) {
  const int tb = blockIdx.x, h = blockIdx.y;
  __shared__ __align__(16) unsigned short ldsA[2][128 * 40];
  __shared__ __align__(16) unsigned short ldsB[2][64 * 40];
  const int tid = threadIdx.x, w = tid >> 5, lane = tid & 31;
  const int ln16 = lane & 15, hi = lane >> 4;
  const int t0 = tb * 128;
  const int mrow = (w & 3) * 32;                           // 32 rows per wave
  const int nh = w >> 2;                                   // 0/1 -> +32 cols
  f32x8 vacc[2][2], eacc[2][2];
#pragma unroll
  for (int mt = 0; mt < 2; ++mt) {
    vacc[mt][0] = zero8(); vacc[mt][1] = zero8();
    eacc[mt][0] = zero8(); eacc[mt][1] = zero8();
  }

  uint2 ar[4];
  float4 br[2];
  auto gload = [&](int it2) {                              // it2 = e*32 + kstep
    int e = it2 >> 5;
    int ks = (it2 & 31) * 32;
    const float* W = Wv + (h * Ee + e) * (Dm * Pp);
#pragma unroll
    for (int it = 0; it < 4; ++it) {
      int idx = tid + it * 256;
      int r = idx >> 3, c4 = (idx & 7) * 4;
      ar[it] = *(const uint2*)&xbf[(t0 + r) * Dm + ks + c4];
    }
#pragma unroll
    for (int it = 0; it < 2; ++it) {
      int idx = tid + it * 256;
      int k = idx >> 4, n4 = (idx & 15) * 4;
      br[it] = *(const float4*)&W[(ks + k) * Pp + n4];
    }
  };
  auto sstore = [&](int buf) {
#pragma unroll
    for (int it = 0; it < 4; ++it) {
      int idx = tid + it * 256;
      int r = idx >> 3, c4 = (idx & 7) * 4;
      *(uint2*)&ldsA[buf][r * 40 + c4] = ar[it];
    }
#pragma unroll
    for (int it = 0; it < 2; ++it) {
      int idx = tid + it * 256;
      int k = idx >> 4, n4 = (idx & 15) * 4;
      float4 v = br[it];
      ldsB[buf][(n4 + 0) * 40 + k] = f2bf(v.x);
      ldsB[buf][(n4 + 1) * 40 + k] = f2bf(v.y);
      ldsB[buf][(n4 + 2) * 40 + k] = f2bf(v.z);
      ldsB[buf][(n4 + 3) * 40 + k] = f2bf(v.w);
    }
  };

  gload(0);
  int cur = 0;
  for (int it2 = 0; it2 < Ee * 32; ++it2) {
    sstore(cur);
    __syncthreads();
    if (it2 + 1 < Ee * 32) gload(it2 + 1);
    bf16x16 a0 = load_a_frag(&ldsA[cur][mrow * 40], 40);
    bf16x16 a1 = load_a_frag(&ldsA[cur][(mrow + 16) * 40], 40);
#pragma unroll
    for (int i = 0; i < 2; ++i) {
      bf16x16 b = load_bt_frag(&ldsB[cur][(nh * 32 + i * 16) * 40], 40, 0);
      eacc[0][i] = wmma_bf16(a0, b, eacc[0][i]);
      eacc[1][i] = wmma_bf16(a1, b, eacc[1][i]);
    }
    cur ^= 1;
    if ((it2 & 31) == 31) {                                // end of expert e
      int e = it2 >> 5;
#pragma unroll
      for (int mt = 0; mt < 2; ++mt)
#pragma unroll
        for (int j = 0; j < 8; ++j) {
          int m = t0 + mrow + mt * 16 + j + hi * 8;
          int b = m >> 11, s = m & (Ss - 1);
          float g = gv[((b * Hh + h) * Ss + s) * Ee + e];
          vacc[mt][0][j] += g * eacc[mt][0][j];
          vacc[mt][1][j] += g * eacc[mt][1][j];
        }
#pragma unroll
      for (int mt = 0; mt < 2; ++mt) { eacc[mt][0] = zero8(); eacc[mt][1] = zero8(); }
    }
  }
#pragma unroll
  for (int mt = 0; mt < 2; ++mt)
#pragma unroll
    for (int i = 0; i < 2; ++i)
#pragma unroll
      for (int j = 0; j < 8; ++j) {
        int m = t0 + mrow + mt * 16 + j + hi * 8;
        int b = m >> 11, s = m & (Ss - 1);
        int p = nh * 32 + i * 16 + ln16;
        vbf[((b * Hh + h) * Ss + s) * Pp + p] = f2bf(vacc[mt][i][j]);
      }
}

// ---------------------------------------------------------------------------
// 5) Flash attention.  grid (16 qblocks, 16 h, 2 b); 128 q rows/WG.
// ---------------------------------------------------------------------------
__global__ void attn_kernel(const unsigned short* __restrict__ qbf,
                            const unsigned short* __restrict__ kbf,
                            const unsigned short* __restrict__ vbf,
                            unsigned short* __restrict__ obf) {
  const int qb = blockIdx.x, h = blockIdx.y, b = blockIdx.z;
  __shared__ __align__(16) unsigned short ldsQ[128 * 72];
  __shared__ __align__(16) unsigned short ldsK[64 * 72];
  __shared__ __align__(16) unsigned short ldsV[64 * 72];     // transposed [p][t]
  __shared__ __align__(16) unsigned short ldsP[8 * 16 * 72]; // per-wave probs
  const int tid = threadIdx.x, w = tid >> 5, lane = tid & 31;
  const int ln16 = lane & 15, hi = lane >> 4;
  const int base = (b * Hh + h) * Ss;
  const int q0 = qb * 128;

#pragma unroll
  for (int it = 0; it < 8; ++it) {                           // Q 128x64 (x4)
    int idx = tid + it * 256;                                // 2048 chunks
    int r = idx >> 4, p4 = (idx & 15) * 4;
    *(uint2*)&ldsQ[r * 72 + p4] =
        *(const uint2*)&qbf[(base + q0 + r) * Pp + p4];
  }
  f32x8 acc[4];
#pragma unroll
  for (int i = 0; i < 4; ++i) acc[i] = zero8();
  float rmax[8], rsum[8];
#pragma unroll
  for (int j = 0; j < 8; ++j) { rmax[j] = -3.0e38f; rsum[j] = 0.0f; }
  unsigned short* Pw = ldsP + w * 16 * 72;

  for (int t0 = 0; t0 < Ss; t0 += 64) {
    __syncthreads();
#pragma unroll
    for (int it = 0; it < 4; ++it) {                         // K & V tiles (x4)
      int idx = tid + it * 256;                              // 1024 chunks
      int t = idx >> 4, p4 = (idx & 15) * 4;
      *(uint2*)&ldsK[t * 72 + p4] =
          *(const uint2*)&kbf[(base + t0 + t) * Pp + p4];
      US4 vv;
      vv.u = *(const uint2*)&vbf[(base + t0 + t) * Pp + p4];
      ldsV[(p4 + 0) * 72 + t] = vv.s[0];
      ldsV[(p4 + 1) * 72 + t] = vv.s[1];
      ldsV[(p4 + 2) * 72 + t] = vv.s[2];
      ldsV[(p4 + 3) * 72 + t] = vv.s[3];
    }
    __syncthreads();
    f32x8 sc[4];
#pragma unroll
    for (int i = 0; i < 4; ++i) sc[i] = zero8();
#pragma unroll
    for (int ks = 0; ks < 64; ks += 32) {
      bf16x16 a = load_a_frag(ldsQ + (w * 16) * 72 + ks, 72);
#pragma unroll
      for (int i = 0; i < 4; ++i) {
        bf16x16 bt = load_bt_frag(ldsK + (i * 16) * 72, 72, ks);
        sc[i] = wmma_bf16(a, bt, sc[i]);
      }
    }
    // online softmax (rows split across 16-lane halves)
#pragma unroll
    for (int j = 0; j < 8; ++j) {
      float rm = -3.0e38f;
#pragma unroll
      for (int i = 0; i < 4; ++i) { sc[i][j] *= 0.125f; rm = fmaxf(rm, sc[i][j]); }
#pragma unroll
      for (int off = 8; off >= 1; off >>= 1) rm = fmaxf(rm, __shfl_xor(rm, off, 32));
      float mnew = fmaxf(rmax[j], rm);
      float corr = __expf(rmax[j] - mnew);
      rmax[j] = mnew;
      rsum[j] *= corr;
#pragma unroll
      for (int i = 0; i < 4; ++i) acc[i][j] *= corr;
      float ps = 0.0f;
#pragma unroll
      for (int i = 0; i < 4; ++i) {
        float pv = __expf(sc[i][j] - mnew);
        sc[i][j] = pv;
        ps += pv;
      }
#pragma unroll
      for (int off = 8; off >= 1; off >>= 1) ps += __shfl_xor(ps, off, 32);
      rsum[j] += ps;
    }
    // D-layout -> A-layout via per-wave LDS (no barrier needed)
#pragma unroll
    for (int i = 0; i < 4; ++i)
#pragma unroll
      for (int j = 0; j < 8; ++j)
        Pw[(j + hi * 8) * 72 + i * 16 + ln16] = f2bf(sc[i][j]);
#pragma unroll
    for (int ks = 0; ks < 64; ks += 32) {
      bf16x16 a = load_a_frag(Pw + ks, 72);
#pragma unroll
      for (int i = 0; i < 4; ++i) {
        bf16x16 bt = load_bt_frag(ldsV + (i * 16) * 72, 72, ks);
        acc[i] = wmma_bf16(a, bt, acc[i]);
      }
    }
  }
#pragma unroll
  for (int j = 0; j < 8; ++j) {
    float invs = 1.0f / rsum[j];
    int r = q0 + w * 16 + j + hi * 8;
#pragma unroll
    for (int i = 0; i < 4; ++i) {
      int p = i * 16 + ln16;
      obf[(base + r) * Pp + p] = f2bf(acc[i][j] * invs);
    }
  }
}

// ---------------------------------------------------------------------------
// 6) Output projection with go gating + residual, pipelined over 128 (h,e):
//    y1[t,d] = src[t,d] + sum_{h,e} go * (o[b,h,s,:] . Wo[h,e,:,d])
//    grid (64 tblocks, 8 dblocks).  Tile 64Mx128N (waves: 4M x 2Nhalf).
// ---------------------------------------------------------------------------
__global__ void oproj_kernel(const unsigned short* __restrict__ obf,
                             const float* __restrict__ go,
                             const float* __restrict__ Wo,
                             const float* __restrict__ src,
                             float* __restrict__ y1) {
  const int tb = blockIdx.x, db = blockIdx.y;
  __shared__ __align__(16) unsigned short ldsA[2][64 * 72];
  __shared__ __align__(16) unsigned short ldsB[2][128 * 72];
  const int tid = threadIdx.x, w = tid >> 5, lane = tid & 31;
  const int ln16 = lane & 15, hi = lane >> 4;
  const int t0 = tb * 64, d0 = db * 128;
  const int mrow = (w & 3) * 16, nh = w >> 2;
  f32x8 acc[4];
#pragma unroll
  for (int i = 0; i < 4; ++i) acc[i] = zero8();

  uint2 oar[4];
  float gar[4];
  float4 wbr[8];
  auto gload = [&](int he) {
    int h = he >> 3, e = he & 7;
#pragma unroll
    for (int it = 0; it < 4; ++it) {                     // o tile + gates
      int idx = tid + it * 256;
      int t = idx >> 4, p4 = (idx & 15) * 4;
      int m = t0 + t;
      int b = m >> 11, s = m & (Ss - 1);
      int bhs = (b * Hh + h) * Ss + s;
      oar[it] = *(const uint2*)&obf[bhs * Pp + p4];
      gar[it] = go[bhs * Ee + e];
    }
    const float* Wp = Wo + (h * Ee + e) * (Pp * Dm);
#pragma unroll
    for (int it = 0; it < 8; ++it) {                     // Wo tile 64x128
      int idx = tid + it * 256;
      int k = idx >> 5, n4 = (idx & 31) * 4;
      wbr[it] = *(const float4*)&Wp[k * Dm + d0 + n4];
    }
  };
  auto sstore = [&](int buf) {
#pragma unroll
    for (int it = 0; it < 4; ++it) {
      int idx = tid + it * 256;
      int t = idx >> 4, p4 = (idx & 15) * 4;
      US4 ov; ov.u = oar[it];
      US4 r;
      r.s[0] = f2bf(bf2f(ov.s[0]) * gar[it]);
      r.s[1] = f2bf(bf2f(ov.s[1]) * gar[it]);
      r.s[2] = f2bf(bf2f(ov.s[2]) * gar[it]);
      r.s[3] = f2bf(bf2f(ov.s[3]) * gar[it]);
      *(uint2*)&ldsA[buf][t * 72 + p4] = r.u;
    }
#pragma unroll
    for (int it = 0; it < 8; ++it) {
      int idx = tid + it * 256;
      int k = idx >> 5, n4 = (idx & 31) * 4;
      float4 v = wbr[it];
      ldsB[buf][(n4 + 0) * 72 + k] = f2bf(v.x);
      ldsB[buf][(n4 + 1) * 72 + k] = f2bf(v.y);
      ldsB[buf][(n4 + 2) * 72 + k] = f2bf(v.z);
      ldsB[buf][(n4 + 3) * 72 + k] = f2bf(v.w);
    }
  };

  gload(0);
  int cur = 0;
  for (int he = 0; he < Hh * Ee; ++he) {
    sstore(cur);
    __syncthreads();
    if (he + 1 < Hh * Ee) gload(he + 1);
#pragma unroll
    for (int ks = 0; ks < 64; ks += 32) {
      bf16x16 a = load_a_frag(&ldsA[cur][mrow * 72 + ks], 72);
#pragma unroll
      for (int i = 0; i < 4; ++i) {
        bf16x16 bt = load_bt_frag(&ldsB[cur][(nh * 64 + i * 16) * 72], 72, ks);
        acc[i] = wmma_bf16(a, bt, acc[i]);
      }
    }
    cur ^= 1;
  }
#pragma unroll
  for (int i = 0; i < 4; ++i)
#pragma unroll
    for (int j = 0; j < 8; ++j) {
      int m = t0 + mrow + j + hi * 8;
      int d = d0 + nh * 64 + i * 16 + ln16;
      y1[m * Dm + d] = src[m * Dm + d] + acc[i][j];
    }
}

// ---------------------------------------------------------------------------
// 7) LayerNorm over last dim (1024). One block per token.
// ---------------------------------------------------------------------------
__global__ void ln_kernel(const float* __restrict__ in, const float* __restrict__ g,
                          const float* __restrict__ bta, float* __restrict__ outF,
                          unsigned short* __restrict__ outBF) {
  __shared__ float red[18];
  int t = blockIdx.x;
  const float* row = in + t * Dm;
  int c4 = threadIdx.x * 4;                      // 256 chunks of 4
  float4 v = *(const float4*)&row[c4];
  float s = v.x + v.y + v.z + v.w;
  float s2 = v.x * v.x + v.y * v.y + v.z * v.z + v.w * v.w;
#pragma unroll
  for (int off = 16; off >= 1; off >>= 1) {
    s += __shfl_xor(s, off, 32);
    s2 += __shfl_xor(s2, off, 32);
  }
  int w = threadIdx.x >> 5, lane = threadIdx.x & 31;
  if (lane == 0) { red[w] = s; red[w + 8] = s2; }
  __syncthreads();
  if (threadIdx.x == 0) {
    float a = 0.0f, c = 0.0f;
    for (int i = 0; i < 8; ++i) { a += red[i]; c += red[i + 8]; }
    red[16] = a; red[17] = c;
  }
  __syncthreads();
  float mean = red[16] * (1.0f / Dm);
  float var = red[17] * (1.0f / Dm) - mean * mean;
  float rstd = rsqrtf(var + 1e-5f);
  float4 gg = *(const float4*)&g[c4];
  float4 bb = *(const float4*)&bta[c4];
  float4 o;
  o.x = (v.x - mean) * rstd * gg.x + bb.x;
  o.y = (v.y - mean) * rstd * gg.y + bb.y;
  o.z = (v.z - mean) * rstd * gg.z + bb.z;
  o.w = (v.w - mean) * rstd * gg.w + bb.w;
  if (outF) *(float4*)&outF[t * Dm + c4] = o;
  if (outBF) {
    US4 r;
    r.s[0] = f2bf(o.x); r.s[1] = f2bf(o.y); r.s[2] = f2bf(o.z); r.s[3] = f2bf(o.w);
    *(uint2*)&outBF[t * Dm + c4] = r.u;
  }
}

// ---------------------------------------------------------------------------
// 8) Generic GEMM (A bf16 in ws, B f32 weights) tile 256Mx64N, double-buffered
//    (8 waves x 32 rows; 8 WMMA per k-step per wave).
// ---------------------------------------------------------------------------
template <bool RELU, bool OUTBF>
__global__ void gemm_kernel(const unsigned short* __restrict__ A,
                            const float* __restrict__ Bw,
                            const float* __restrict__ bias,
                            const float* __restrict__ resid,
                            unsigned short* __restrict__ outBF,
                            float* __restrict__ outF, int K, int N) {
  const int mb = blockIdx.x, nb = blockIdx.y;
  __shared__ __align__(16) unsigned short ldsA[2][256 * 40];
  __shared__ __align__(16) unsigned short ldsB[2][64 * 40];
  const int tid = threadIdx.x, w = tid >> 5, lane = tid & 31;
  const int ln16 = lane & 15, hi = lane >> 4;
  const int m0 = mb * 256, n0 = nb * 64;
  f32x8 acc[2][4];
#pragma unroll
  for (int mt = 0; mt < 2; ++mt)
#pragma unroll
    for (int i = 0; i < 4; ++i) acc[mt][i] = zero8();

  uint2 ar[8];
  float4 br[2];
  auto gload = [&](int ks) {
#pragma unroll
    for (int it = 0; it < 8; ++it) {                  // A 256x32 (bf16 copy)
      int idx = tid + it * 256;
      int r = idx >> 3, c4 = (idx & 7) * 4;
      ar[it] = *(const uint2*)&A[(m0 + r) * K + ks + c4];
    }
#pragma unroll
    for (int it = 0; it < 2; ++it) {                  // B 32x64
      int idx = tid + it * 256;
      int k = idx >> 4, n4 = (idx & 15) * 4;
      br[it] = *(const float4*)&Bw[(ks + k) * N + n0 + n4];
    }
  };
  auto sstore = [&](int buf) {
#pragma unroll
    for (int it = 0; it < 8; ++it) {
      int idx = tid + it * 256;
      int r = idx >> 3, c4 = (idx & 7) * 4;
      *(uint2*)&ldsA[buf][r * 40 + c4] = ar[it];
    }
#pragma unroll
    for (int it = 0; it < 2; ++it) {
      int idx = tid + it * 256;
      int k = idx >> 4, n4 = (idx & 15) * 4;
      float4 v = br[it];
      ldsB[buf][(n4 + 0) * 40 + k] = f2bf(v.x);
      ldsB[buf][(n4 + 1) * 40 + k] = f2bf(v.y);
      ldsB[buf][(n4 + 2) * 40 + k] = f2bf(v.z);
      ldsB[buf][(n4 + 3) * 40 + k] = f2bf(v.w);
    }
  };

  gload(0);
  int cur = 0;
  for (int ks = 0; ks < K; ks += 32) {
    sstore(cur);
    __syncthreads();
    if (ks + 32 < K) gload(ks + 32);
    bf16x16 a0 = load_a_frag(&ldsA[cur][(w * 32) * 40], 40);
    bf16x16 a1 = load_a_frag(&ldsA[cur][(w * 32 + 16) * 40], 40);
#pragma unroll
    for (int i = 0; i < 4; ++i) {
      bf16x16 b = load_bt_frag(&ldsB[cur][(i * 16) * 40], 40, 0);
      acc[0][i] = wmma_bf16(a0, b, acc[0][i]);
      acc[1][i] = wmma_bf16(a1, b, acc[1][i]);
    }
    cur ^= 1;
  }
#pragma unroll
  for (int mt = 0; mt < 2; ++mt)
#pragma unroll
    for (int i = 0; i < 4; ++i)
#pragma unroll
      for (int j = 0; j < 8; ++j) {
        int m = m0 + w * 32 + mt * 16 + j + hi * 8;
        int n = n0 + i * 16 + ln16;
        float v = acc[mt][i][j] + bias[n];
        if (RELU) v = fmaxf(v, 0.0f);
        if (OUTBF) outBF[m * N + n] = f2bf(v);
        else       outF[m * N + n] = resid[m * N + n] + v;
      }
}

// ---------------------------------------------------------------------------
// Host-side launch
// ---------------------------------------------------------------------------
extern "C" void kernel_launch(void* const* d_in, const int* in_sizes, int n_in,
                              void* d_out, int out_size, void* d_ws, size_t ws_size,
                              hipStream_t stream) {
  (void)in_sizes; (void)n_in; (void)out_size; (void)ws_size;
  const float* src    = (const float*)d_in[0];
  const float* Wq     = (const float*)d_in[1];
  const float* Wk     = (const float*)d_in[2];
  const float* Wv     = (const float*)d_in[3];
  const float* Wo     = (const float*)d_in[4];
  const float* Wsel_v = (const float*)d_in[5];
  const float* Wsel_o = (const float*)d_in[6];
  const float* ln1_g  = (const float*)d_in[7];
  const float* ln1_b  = (const float*)d_in[8];
  const float* ln2_g  = (const float*)d_in[9];
  const float* ln2_b  = (const float*)d_in[10];
  const float* W1     = (const float*)d_in[11];
  const float* b1     = (const float*)d_in[12];
  const float* W2     = (const float*)d_in[13];
  const float* b2     = (const float*)d_in[14];
  float* out = (float*)d_out;

  char* ws = (char*)d_ws;
  size_t off = 0;
  unsigned short* qbf = (unsigned short*)(ws + off); off += (size_t)Bb * Hh * Ss * Pp * 2;  // 8MB
  unsigned short* kbf = (unsigned short*)(ws + off); off += (size_t)Bb * Hh * Ss * Pp * 2;  // 8MB
  unsigned short* vbf = (unsigned short*)(ws + off); off += (size_t)Bb * Hh * Ss * Pp * 2;  // 8MB
  unsigned short* obf = (unsigned short*)(ws + off); off += (size_t)Bb * Hh * Ss * Pp * 2;  // 8MB
  float* logits = (float*)(ws + off); off += (size_t)2 * Bb * Hh * Ss * Ee * 4;             // 4MB
  float* gv     = (float*)(ws + off); off += (size_t)Bb * Hh * Ss * Ee * 4;                 // 2MB
  float* go     = (float*)(ws + off); off += (size_t)Bb * Hh * Ss * Ee * 4;                 // 2MB
  float* y1     = (float*)(ws + off); off += (size_t)Mtok * Dm * 4;                         // 16MB
  float* xnf    = (float*)(ws + off); off += (size_t)Mtok * Dm * 4;                         // 16MB
  unsigned short* xnbf = (unsigned short*)(ws + off); off += (size_t)Mtok * Dm * 2;         // 8MB
  unsigned short* hbuf = (unsigned short*)(ws + off); off += (size_t)Mtok * Ff * 2;         // 32MB
  unsigned short* xbf  = (unsigned short*)(ws + off); off += (size_t)Mtok * Dm * 2;         // 8MB
  float* y2 = y1;  // y1 is dead after LN1; reuse for FFN2 output

  tobf_kernel<<<Mtok * Dm / 1024, 256, 0, stream>>>(src, xbf);
  gate_logits_kernel<<<Mtok, 256, 0, stream>>>(src, Wsel_v, Wsel_o, logits);
  gate_topk_kernel<<<(2 * Bb * Hh * Ss) / 256, 256, 0, stream>>>(logits, gv, go);
  qk_rope_kernel<<<dim3(Mtok / 128, Hh, 2), 256, 0, stream>>>(xbf, Wq, Wk, qbf, kbf);
  vexpert_kernel<<<dim3(Mtok / 128, Hh), 256, 0, stream>>>(xbf, Wv, gv, vbf);
  attn_kernel<<<dim3(Ss / 128, Hh, Bb), 256, 0, stream>>>(qbf, kbf, vbf, obf);
  oproj_kernel<<<dim3(Mtok / 64, Dm / 128), 256, 0, stream>>>(obf, go, Wo, src, y1);
  ln_kernel<<<Mtok, 256, 0, stream>>>(y1, ln1_g, ln1_b, xnf, xnbf);
  gemm_kernel<true, true><<<dim3(Mtok / 256, Ff / 64), 256, 0, stream>>>(
      xnbf, W1, b1, nullptr, hbuf, nullptr, Dm, Ff);
  gemm_kernel<false, false><<<dim3(Mtok / 256, Dm / 64), 256, 0, stream>>>(
      hbuf, W2, b2, xnf, nullptr, y2, Ff, Dm);
  ln_kernel<<<Mtok, 256, 0, stream>>>(y2, ln2_g, ln2_b, out, nullptr);
}
